// LocalDecoder_CBatchNorm_27470610825331
// MI455X (gfx1250) — matrix operand order repaired and checked by reading, not compiled, not run on hardware
//
#include <hip/hip_runtime.h>
#include <hip/hip_bf16.h>

typedef __attribute__((ext_vector_type(16))) _Float16 v16h;
typedef __attribute__((ext_vector_type(8)))  _Float16 v8h;
typedef __attribute__((ext_vector_type(8)))  float    v8f;

#define T_PTS   32768
#define TILE    128
#define ROWH    264                 // padded LDS row (halves): 264*2B = 528B (16B-multiple, bank-skewed)
#define MAT_H   65536               // halves per swizzled 256x256 weight matrix
#define FCOUT_OFF (32 * 65536)      // swizzled fc_out starts here (halves)
#define SMEM_BYTES (2 * TILE * ROWH * 2 + TILE * 2 * 4)

struct WSrc { const float* w[33]; };

struct Args {
  const float* p;
  const float* c_global;
  const float* c_local;
  const _Float16* wt;               // swizzled f16 weights in workspace
  const float* bias[32];            // per blk {g0,b0,fc0,g1,b1,fc1}, 30=bn_gamma, 31=bn_beta
  const float* fcout_b;
  const float* fcp_w;
  const float* fcp_b;
  float* out;
};

// ---------------- weight pre-swizzle: f32 [N=256][K=256] -> f16 WMMA-B-fragment order ----------------
// flat index within a matrix: ((nt*8 + k)*32 + lane)*16 + h
//   B-frag layout (16-bit B, 32x16 KxN): lane<16 -> N=lane,    K = k*32 + {0..7, 16..23}[h]
//                                        lane>=16 -> N=lane-16, K = k*32 + {8..15, 24..31}[h]
__global__ __launch_bounds__(256) void prep_weights(WSrc ws, _Float16* __restrict__ dst) {
  int id = blockIdx.x * 256 + threadIdx.x;
  int e, m;
  const float* src;
  if (id < 32 * 65536) {
    m = id >> 16; e = id & 65535; src = ws.w[m];
  } else if (id < 32 * 65536 + 8192) {
    e = id - 32 * 65536; src = ws.w[32];         // fc_out [32][256]
  } else {
    return;
  }
  int h    = e & 15;
  int lane = (e >> 4) & 31;
  int nt   = e >> 12;
  int kk   = (e >> 9) & 7;
  int n    = nt * 16 + (lane & 15);
  int k    = kk * 32 + ((lane >> 4) << 3) + (h & 7) + ((h >> 3) << 4);
  dst[id] = (_Float16)src[n * 256 + k];
}

// ---------------- per-wave WMMA GEMM: acc[16][8] += A(16x256 from LDS rows) * W(256x256 swizzled) ----
__device__ __forceinline__ void gemm_acc(v8f (&acc)[16], const _Float16* sX,
                                         const _Float16* __restrict__ wt,
                                         int laneM, int c0, int lane) {
#pragma unroll 1
  for (int k = 0; k < 8; ++k) {
    // A fragment (16x32 f16): lane<16 -> M=lane, K in {c..c+7} U {c+16..c+23} with c=k*32 (+8 for hi lanes)
    const _Float16* ar = sX + laneM * ROWH + k * 32 + c0;
    v8h lo = *(const v8h*)ar;
    v8h hi = *(const v8h*)(ar + 16);
    v16h a = __builtin_shufflevector(lo, hi, 0,1,2,3,4,5,6,7,8,9,10,11,12,13,14,15);
    const _Float16* wk = wt + (k * 32 + lane) * 16;
#pragma unroll
    for (int nt = 0; nt < 16; ++nt) {
      v16h bf = *(const v16h*)(wk + nt * 4096);
      acc[nt] = __builtin_amdgcn_wmma_f32_16x16x32_f16(false, a, false, bf,
                                                       (short)0, acc[nt], false, false);
    }
  }
}

__device__ __forceinline__ void bias_set(v8f (&acc)[16], const float* __restrict__ b, int nLo) {
#pragma unroll
  for (int nt = 0; nt < 16; ++nt) {
    float v = b[nt * 16 + nLo];
#pragma unroll
    for (int j = 0; j < 8; ++j) acc[nt][j] = v;
  }
}
__device__ __forceinline__ void bias_add(v8f (&acc)[16], const float* __restrict__ b, int nLo) {
#pragma unroll
  for (int nt = 0; nt < 16; ++nt) {
    float v = b[nt * 16 + nLo];
#pragma unroll
    for (int j = 0; j < 8; ++j) acc[nt][j] += v;
  }
}
__device__ __forceinline__ void mul_net(v8f (&acc)[16], const v8f (&m)[16], float s) {
#pragma unroll
  for (int nt = 0; nt < 16; ++nt)
#pragma unroll
    for (int j = 0; j < 8; ++j) acc[nt][j] *= m[nt][j] * s;
}
__device__ __forceinline__ void add_acc(v8f (&d)[16], const v8f (&s)[16]) {
#pragma unroll
  for (int nt = 0; nt < 16; ++nt)
#pragma unroll
    for (int j = 0; j < 8; ++j) d[nt][j] += s[nt][j];
}
// relu + f16, write wave-private rows into LDS staging tile (no barrier needed: same-wave reuse only)
__device__ __forceinline__ void relu_store(const v8f (&acc)[16], _Float16* sT,
                                           int ptBase, int hi8, int nLo) {
#pragma unroll
  for (int nt = 0; nt < 16; ++nt)
#pragma unroll
    for (int j = 0; j < 8; ++j) {
      float v = acc[nt][j];
      v = v > 0.f ? v : 0.f;
      sT[(ptBase + j + hi8) * ROWH + nt * 16 + nLo] = (_Float16)v;
    }
}

__device__ __forceinline__ void res_block(v8f (&net)[16], const Args& args,
                                          const _Float16* sC, _Float16* sT, int s0,
                                          int laneM, int c0, int lane,
                                          int ptBase, int hi8, int nLo, float inv) {
  const _Float16* wt = args.wt;
  v8f acc[16], acc2[16];
  // t = relu(cbn(net)) ; CBN fused: acc = (c@Wg + bg)*net*inv, then GEMM-accumulate (c@Wb + bb)
  bias_set(acc, args.bias[s0 + 0], nLo);
  gemm_acc(acc, sC, wt + (size_t)(s0 + 0) * MAT_H, laneM, c0, lane);
  mul_net(acc, net, inv);
  bias_add(acc, args.bias[s0 + 1], nLo);
  gemm_acc(acc, sC, wt + (size_t)(s0 + 1) * MAT_H, laneM, c0, lane);
  relu_store(acc, sT, ptBase, hi8, nLo);
  // h = t @ fc0^T + b
  bias_set(acc, args.bias[s0 + 2], nLo);
  gemm_acc(acc, sT, wt + (size_t)(s0 + 2) * MAT_H, laneM, c0, lane);   // h lives in acc
  // dx = relu(cbn(h)) @ fc1^T + b
  bias_set(acc2, args.bias[s0 + 3], nLo);
  gemm_acc(acc2, sC, wt + (size_t)(s0 + 3) * MAT_H, laneM, c0, lane);
  mul_net(acc2, acc, inv);
  bias_add(acc2, args.bias[s0 + 4], nLo);
  gemm_acc(acc2, sC, wt + (size_t)(s0 + 4) * MAT_H, laneM, c0, lane);
  relu_store(acc2, sT, ptBase, hi8, nLo);
  bias_set(acc2, args.bias[s0 + 5], nLo);
  gemm_acc(acc2, sT, wt + (size_t)(s0 + 5) * MAT_H, laneM, c0, lane);
  add_acc(net, acc2);                                                  // net += dx
}

// ---------------- fused decoder: one tile of 128 points per workgroup, net resident in VGPRs --------
__global__ __launch_bounds__(256) void decoder_kernel(Args args) {
  extern __shared__ char smem[];
  _Float16* sC = (_Float16*)smem;                 // [TILE][ROWH] f16 conditioning matrix
  _Float16* sT = sC + TILE * ROWH;                // [TILE][ROWH] f16 activation staging
  float*    sP = (float*)(sT + TILE * ROWH);      // [TILE][2]   point coords

  const int tid  = threadIdx.x;
  const int lane = tid & 31;
  const int wave = tid >> 5;
  const int tile = blockIdx.x;
  const int b    = tile >> 8;                     // 256 tiles per batch (T/TILE)
  const int t0   = (tile & 255) * TILE;

  // ---- Phase A: build c-tile cooperatively ----
  sP[tid] = args.p[((size_t)b * T_PTS + t0) * 2 + tid];   // 128 pts * 2 floats = 256
  {
    const int ch = tid & 127;
    float cgv = args.c_global[b * 128 + ch];
#pragma unroll 1
    for (int pt = tid >> 7; pt < TILE; pt += 2)
      sC[pt * ROWH + ch] = (_Float16)cgv;
  }
  __syncthreads();
  {
    const int ch = tid & 127;
    const float* fc = args.c_local + ((size_t)b * 128 + ch) * (128 * 128);
#pragma unroll 1
    for (int pt = tid >> 7; pt < TILE; pt += 2) {
      float px = sP[pt * 2 + 0], py = sP[pt * 2 + 1];
      // normalize_coordinate + align_corners grid math: clip(((p/1.101)+0.5) * 127)
      float x = px * (1.0f / 1.101f) + 0.5f;
      float y = py * (1.0f / 1.101f) + 0.5f;
      x = fminf(fmaxf(x * 127.0f, 0.0f), 127.0f);
      y = fminf(fmaxf(y * 127.0f, 0.0f), 127.0f);
      float x0f = floorf(x), y0f = floorf(y);
      int x0 = (int)x0f, y0 = (int)y0f;
      int x1 = x0 + 1 < 127 ? x0 + 1 : 127;
      int y1 = y0 + 1 < 127 ? y0 + 1 : 127;
      float wx = x - x0f, wy = y - y0f;
      float f00 = fc[y0 * 128 + x0], f01 = fc[y0 * 128 + x1];
      float f10 = fc[y1 * 128 + x0], f11 = fc[y1 * 128 + x1];
      float v = f00 * (1 - wx) * (1 - wy) + f01 * wx * (1 - wy)
              + f10 * (1 - wx) * wy + f11 * wx * wy;
      sC[pt * ROWH + 128 + ch] = (_Float16)v;
    }
  }
  __syncthreads();

  // ---- Phase B: per-wave 16-row GEMM chain ----
  const int ptBase = wave * 16;
  const int laneM  = ptBase + (lane & 15);        // A-matrix row this lane feeds
  const int c0     = (lane >> 4) * 8;             // A-frag K sub-offset
  const int hi8    = (lane >> 4) * 8;             // C/D M offset for hi half-wave
  const int nLo    = lane & 15;                   // C/D N within tile
  const float inv  = rsqrtf(1.0f + 1e-5f);        // eval-mode BN scale

  v8f net[16];
  {
    // net = p @ fc_p^T + b  (K=2, do in VALU)
#pragma unroll
    for (int nt = 0; nt < 16; ++nt) {
      int n = nt * 16 + nLo;
      float w0 = args.fcp_w[n * 2 + 0], w1 = args.fcp_w[n * 2 + 1], bb = args.fcp_b[n];
#pragma unroll
      for (int j = 0; j < 8; ++j) {
        int pt = ptBase + j + hi8;
        net[nt][j] = w0 * sP[pt * 2 + 0] + w1 * sP[pt * 2 + 1] + bb;
      }
    }
  }

  res_block(net, args, sC, sT, 0,  laneM, c0, lane, ptBase, hi8, nLo, inv);
  res_block(net, args, sC, sT, 6,  laneM, c0, lane, ptBase, hi8, nLo, inv);
  res_block(net, args, sC, sT, 12, laneM, c0, lane, ptBase, hi8, nLo, inv);
  res_block(net, args, sC, sT, 18, laneM, c0, lane, ptBase, hi8, nLo, inv);
  res_block(net, args, sC, sT, 24, laneM, c0, lane, ptBase, hi8, nLo, inv);

  // final CBN + relu
  {
    v8f acc[16];
    bias_set(acc, args.bias[30], nLo);
    gemm_acc(acc, sC, args.wt + (size_t)30 * MAT_H, laneM, c0, lane);
    mul_net(acc, net, inv);
    bias_add(acc, args.bias[31], nLo);
    gemm_acc(acc, sC, args.wt + (size_t)31 * MAT_H, laneM, c0, lane);
    relu_store(acc, sT, ptBase, hi8, nLo);
  }

  // out = t @ fc_out^T + b   (N=32 -> two 16-wide tiles)
  {
    v8f o0, o1;
    float ob0 = args.fcout_b[nLo], ob1 = args.fcout_b[16 + nLo];
#pragma unroll
    for (int j = 0; j < 8; ++j) { o0[j] = ob0; o1[j] = ob1; }
    const _Float16* wo = args.wt + FCOUT_OFF;
#pragma unroll 1
    for (int k = 0; k < 8; ++k) {
      const _Float16* ar = sT + laneM * ROWH + k * 32 + c0;
      v8h lo = *(const v8h*)ar;
      v8h hi = *(const v8h*)(ar + 16);
      v16h a = __builtin_shufflevector(lo, hi, 0,1,2,3,4,5,6,7,8,9,10,11,12,13,14,15);
      v16h b0 = *(const v16h*)(wo + (k * 32 + lane) * 16);
      v16h b1 = *(const v16h*)(wo + 4096 + (k * 32 + lane) * 16);
      o0 = __builtin_amdgcn_wmma_f32_16x16x32_f16(false, a, false, b0, (short)0, o0, false, false);
      o1 = __builtin_amdgcn_wmma_f32_16x16x32_f16(false, a, false, b1, (short)0, o1, false, false);
    }
    const size_t rowBase = (size_t)b * T_PTS + t0;
#pragma unroll
    for (int j = 0; j < 8; ++j) {
      int pt = ptBase + j + hi8;
      size_t off = (rowBase + pt) * 32;
      args.out[off + nLo]      = o0[j];
      args.out[off + 16 + nLo] = o1[j];
    }
  }
}

extern "C" void kernel_launch(void* const* d_in, const int* in_sizes, int n_in,
                              void* d_out, int out_size, void* d_ws, size_t ws_size,
                              hipStream_t stream) {
  // JAX pytree (alphabetical) flatten order:
  // 0:c_global 1:c_local 2:p
  // then params.blocks[0..4], each (alphabetical): b0.{b,w} b1.{b,w} fc0.{b,w} fc1.{b,w} g0.{b,w} g1.{b,w}
  // then bn_beta.{b,w} bn_gamma.{b,w} fc_out.{b,w} fc_p.{b,w}
  (void)in_sizes; (void)n_in; (void)out_size;

  WSrc wsrc;
  Args args;
  args.c_global = (const float*)d_in[0];
  args.c_local  = (const float*)d_in[1];
  args.p        = (const float*)d_in[2];

  for (int blk = 0; blk < 5; ++blk) {
    int base = 3 + blk * 12;
    // stage order: g0, b0, fc0, g1, b1, fc1
    wsrc.w[6 * blk + 0] = (const float*)d_in[base + 9];   // g0.w
    wsrc.w[6 * blk + 1] = (const float*)d_in[base + 1];   // b0.w
    wsrc.w[6 * blk + 2] = (const float*)d_in[base + 5];   // fc0.w
    wsrc.w[6 * blk + 3] = (const float*)d_in[base + 11];  // g1.w
    wsrc.w[6 * blk + 4] = (const float*)d_in[base + 3];   // b1.w
    wsrc.w[6 * blk + 5] = (const float*)d_in[base + 7];   // fc1.w
    args.bias[6 * blk + 0] = (const float*)d_in[base + 8];   // g0.b
    args.bias[6 * blk + 1] = (const float*)d_in[base + 0];   // b0.b
    args.bias[6 * blk + 2] = (const float*)d_in[base + 4];   // fc0.b
    args.bias[6 * blk + 3] = (const float*)d_in[base + 10];  // g1.b
    args.bias[6 * blk + 4] = (const float*)d_in[base + 2];   // b1.b
    args.bias[6 * blk + 5] = (const float*)d_in[base + 6];   // fc1.b
  }
  wsrc.w[30] = (const float*)d_in[66];   // bn_gamma.w
  wsrc.w[31] = (const float*)d_in[64];   // bn_beta.w
  wsrc.w[32] = (const float*)d_in[68];   // fc_out.w
  args.bias[30] = (const float*)d_in[65]; // bn_gamma.b
  args.bias[31] = (const float*)d_in[63]; // bn_beta.b
  args.fcout_b  = (const float*)d_in[67];
  args.fcp_b    = (const float*)d_in[69];
  args.fcp_w    = (const float*)d_in[70];

  _Float16* wt = (_Float16*)d_ws;        // needs 32*65536+8192 halves ~ 4.02 MB
  args.wt  = wt;
  args.out = (float*)d_out;

  // 1) swizzle weights into f16 WMMA-fragment layout (L2-resident thereafter)
  prep_weights<<<dim3((32 * 65536 + 8192) / 256), dim3(256), 0, stream>>>(wsrc, wt);
  // 2) fused decoder: 4 batches * 256 tiles of 128 points
  decoder_kernel<<<dim3(4 * 256), dim3(256), SMEM_BYTES, stream>>>(args);
}
